// RWKV7_22342419873910
// MI455X (gfx1250) — matrix operand order repaired and checked by reading, not compiled, not run on hardware
//
#include <hip/hip_runtime.h>
#include <math.h>

// ---------------- problem constants ----------------
static constexpr int kB = 4;
static constexpr int kT = 1024;
static constexpr int kC = 768;
static constexpr int kH = 12;
static constexpr int kN = 64;
static constexpr int kBT = kB * kT;            // 4096 rows
static constexpr float kGnEps = 6.4e-4f;

// ---------------- WMMA types -----------------------
typedef __attribute__((ext_vector_type(16))) __bf16 v16bf;
typedef __attribute__((ext_vector_type(8)))  float  v8f;

__device__ __forceinline__ __bf16 bf_from_bits(unsigned short u) {
    union { unsigned short s; __bf16 b; } x; x.s = u; return x.b;
}

// ---------------- generic bf16 WMMA GEMM (TN) -----------
// C[M,N] = act(A[M,K] @ Bt[N,K]^T); fp32 in memory, bf16 WMMA compute.
// REQUIRES: M % 128 == 0, N % 64 == 0, K % 32 == 0 (weights pre-padded).
// Block tile 128x64, 8 waves, each wave owns 32x32 = 4 WMMA accumulators.
// LDS double-buffered: global loads for tile i+1 issue before the barrier,
// waits overlap the WMMAs of tile i.
#define GBM 128
#define GBN 64
#define GBK 32

__global__ __launch_bounds__(256)
void gemm_wmma_bf16(const float* __restrict__ A, int lda,
                    const float* __restrict__ Bt, int ldbt,
                    float* __restrict__ Cm, int ldc,
                    int M, int N, int K, int act)
{
    __shared__ __bf16 As[2][GBM * GBK];   // 2 x 8 KB, [m][k]
    __shared__ __bf16 Bs[2][GBN * GBK];   // 2 x 4 KB, [n][k]

    const int tid  = threadIdx.x;
    const int lane = tid & 31;            // wave32
    const int wave = tid >> 5;            // 0..7
    const int lo   = lane & 15;
    const int hi   = (lane >> 4) & 1;
    const int wr   = (wave >> 1) * 32;    // wave row offset in tile
    const int wc   = (wave & 1) * 32;     // wave col offset in tile
    const int m0   = blockIdx.y * GBM;
    const int n0   = blockIdx.x * GBN;

    float4 fa[4], fb[2];

    auto gload = [&](int k0) {            // issue all global loads first
        #pragma unroll
        for (int it = 0; it < 4; ++it) {
            int i = (it * 256 + tid) * 4;          // 0..4095
            int r = i >> 5, c = i & 31;
            fa[it] = *(const float4*)(A + (size_t)(m0 + r) * lda + k0 + c);
        }
        #pragma unroll
        for (int it = 0; it < 2; ++it) {
            int i = (it * 256 + tid) * 4;          // 0..2047
            int r = i >> 5, c = i & 31;
            fb[it] = *(const float4*)(Bt + (size_t)(n0 + r) * ldbt + k0 + c);
        }
    };
    auto lstore = [&](int buf) {          // fp32 -> bf16 pack, b64 LDS stores
        #pragma unroll
        for (int it = 0; it < 4; ++it) {
            int i = (it * 256 + tid) * 4;
            __bf16 t[4] = { (__bf16)fa[it].x, (__bf16)fa[it].y,
                            (__bf16)fa[it].z, (__bf16)fa[it].w };
            *(uint2*)&As[buf][i] = *(const uint2*)t;
        }
        #pragma unroll
        for (int it = 0; it < 2; ++it) {
            int i = (it * 256 + tid) * 4;
            __bf16 t[4] = { (__bf16)fb[it].x, (__bf16)fb[it].y,
                            (__bf16)fb[it].z, (__bf16)fb[it].w };
            *(uint2*)&Bs[buf][i] = *(const uint2*)t;
        }
    };

    v8f acc[2][2] = {};

    gload(0);
    lstore(0);

    for (int k0 = 0; k0 < K; k0 += GBK) {
        const int  cur = (k0 >> 5) & 1;
        const bool nx  = (k0 + GBK) < K;
        if (nx) gload(k0 + GBK);          // loads in flight across barrier
        __syncthreads();                  // cur buffer ready block-wide

        // ---- fragments (ISA 16-bit A/B layouts, pairwise b32 LDS reads) ----
        const unsigned* As32 = (const unsigned*)As[cur];
        const unsigned* Bs32 = (const unsigned*)Bs[cur];
        v16bf afr[2], bfr[2];
        #pragma unroll
        for (int s = 0; s < 2; ++s) {
            int arow = wr + s * 16 + lo;           // A: M = lane lo
            int bcol = wc + s * 16 + lo;           // B: N = lane lo
            #pragma unroll
            for (int v = 0; v < 8; ++v) {
                // A K index: v<4 -> 8*hi+2v ; v>=4 -> 16+8*hi+2(v-4)
                int kla = (v < 4) ? (8 * hi + 2 * v) : (16 + 8 * hi + 2 * (v - 4));
                unsigned wa = As32[(arow * GBK + kla) >> 1];
                afr[s][2 * v]     = bf_from_bits((unsigned short)(wa & 0xffffu));
                afr[s][2 * v + 1] = bf_from_bits((unsigned short)(wa >> 16));
                // B K index: same as A
                int klb = (v < 4) ? (8 * hi + 2 * v) : (16 + 8 * hi + 2 * (v - 4));
                unsigned wb = Bs32[(bcol * GBK + klb) >> 1];
                bfr[s][2 * v]     = bf_from_bits((unsigned short)(wb & 0xffffu));
                bfr[s][2 * v + 1] = bf_from_bits((unsigned short)(wb >> 16));
            }
        }
        #pragma unroll
        for (int i2 = 0; i2 < 2; ++i2)
            #pragma unroll
            for (int j2 = 0; j2 < 2; ++j2)
                acc[i2][j2] = __builtin_amdgcn_wmma_f32_16x16x32_bf16(
                    false, afr[i2], false, bfr[j2], (short)0, acc[i2][j2],
                    false, false);

        if (nx) lstore(cur ^ 1);          // fill other buffer (no race: barrier)
    }

    // C/D layout: VGPR v -> row (v + 8*hi), col = lane lo
    #pragma unroll
    for (int i2 = 0; i2 < 2; ++i2) {
        #pragma unroll
        for (int j2 = 0; j2 < 2; ++j2) {
            int col = n0 + wc + j2 * 16 + lo;
            #pragma unroll
            for (int vq = 0; vq < 8; ++vq) {
                int row = m0 + wr + i2 * 16 + vq + 8 * hi;
                float o = acc[i2][j2][vq];
                if (act == 1)      o = tanhf(o);
                else if (act == 2) o = 1.0f / (1.0f + expf(-o));
                Cm[(size_t)row * ldc + col] = o;
            }
        }
    }
}

// ---------------- weight transpose + zero-pad -------------------
// src is K0 x N0 (row-major). dst is Np x Kp with dst[n*Kp+k] = src[k*N0+n],
// zero outside.
__global__ __launch_bounds__(256)
void transpose_pad_kernel(const float* __restrict__ src, int K0, int N0,
                          float* __restrict__ dst, int Np, int Kp)
{
    int idx = blockIdx.x * 256 + threadIdx.x;
    if (idx >= Np * Kp) return;
    int n = idx / Kp, k = idx % Kp;
    dst[idx] = (n < N0 && k < K0) ? src[(size_t)k * N0 + n] : 0.0f;
}

// maa_w1 (768x112, 4 chunks of 28) -> transposed 128x768, chunk f occupies
// output rows 32f..32f+27 (rows 32f+28..32f+31 zero).
__global__ __launch_bounds__(256)
void trans_maa_w1_kernel(const float* __restrict__ src, float* __restrict__ dst)
{
    int idx = blockIdx.x * 256 + threadIdx.x;
    if (idx >= 128 * kC) return;
    int n = idx / kC, k = idx % kC;
    int f = n >> 5, j = n & 31;
    dst[idx] = (j < 28) ? src[(size_t)k * 112 + f * 28 + j] : 0.0f;
}

// ---------------- elementwise kernels ---------------
__device__ __forceinline__ float sigm(float x) { return 1.0f / (1.0f + expf(-x)); }

__device__ __forceinline__ float warp_sum(float v) {
    #pragma unroll
    for (int m = 16; m > 0; m >>= 1) v += __shfl_xor(v, m, 32);
    return v;
}

// token shift: xx = x[t-1]-x[t] (zero pad at t=0); xxx = x + xx*time_maa_x
__global__ __launch_bounds__(256)
void prelude_kernel(const float* __restrict__ x, const float* __restrict__ tmx,
                    float* __restrict__ xx, float* __restrict__ xxx, int total)
{
    int idx = blockIdx.x * 256 + threadIdx.x;
    if (idx >= total) return;
    int c  = idx % kC;
    int bt = idx / kC;
    int t  = bt % kT;
    float xc = x[idx];
    float xp = (t == 0) ? 0.0f : x[idx - kC];
    float d  = xp - xc;
    xx[idx]  = d;
    xxx[idx] = xc + d * tmx[c];
}

// x{rg,wa,k,v} = x + xx * (time_maa_* + adds_f)
__global__ __launch_bounds__(256)
void mix4_kernel(const float* __restrict__ x, const float* __restrict__ xx,
                 const float* __restrict__ a0, const float* __restrict__ a1,
                 const float* __restrict__ a2, const float* __restrict__ a3,
                 const float* __restrict__ trg, const float* __restrict__ twa,
                 const float* __restrict__ tk,  const float* __restrict__ tv,
                 float* __restrict__ xrg, float* __restrict__ xwa,
                 float* __restrict__ xk,  float* __restrict__ xv, int total)
{
    int idx = blockIdx.x * 256 + threadIdx.x;
    if (idx >= total) return;
    int c = idx % kC;
    float xc = x[idx], d = xx[idx];
    xrg[idx] = xc + d * (trg[c] + a0[idx]);
    xwa[idx] = xc + d * (twa[c] + a1[idx]);
    xk[idx]  = xc + d * (tk[c]  + a2[idx]);
    xv[idx]  = xc + d * (tv[c]  + a3[idx]);
}

// Fused pre-scan math: one warp per head, 2 channels per lane. In-place:
//   wtmp -> d = exp(-exp(w)), atmp -> a, kktmp -> kk(normed), kraw -> k, vraw -> v
__global__ __launch_bounds__(384)
void fuse_pre_scan(float* __restrict__ wtmp_d, float* __restrict__ atmp_a,
                   float* __restrict__ kktmp_kk, const float* __restrict__ matmp,
                   const float* __restrict__ mktmp, const float* __restrict__ mvtmp,
                   float* __restrict__ kraw_k, float* __restrict__ vraw_v,
                   const float* __restrict__ v1,
                   const float* __restrict__ tdecay, const float* __restrict__ taaaaa,
                   const float* __restrict__ tmisca, const float* __restrict__ tmisck,
                   const float* __restrict__ tmiscv)
{
    int bt = blockIdx.x;
    int h  = threadIdx.x >> 5;
    int l  = threadIdx.x & 31;
    size_t base = (size_t)bt * kC + h * kN;
    size_t ii[2] = { base + l, base + l + 32 };
    int    cc[2] = { h * kN + l, h * kN + l + 32 };
    float od[2], oa[2], ok[2], ov[2], kkun[2];

    #pragma unroll
    for (int q = 0; q < 2; ++q) {
        size_t i = ii[q]; int c = cc[q];
        float wv = tdecay[c] + wtmp_d[i];
        wv = -log1pf(expf(-wv)) - 0.5f;          // -softplus(-x) - 0.5
        od[q] = expf(-expf(wv));
        float a   = sigm(taaaaa[c] + atmp_a[i]);
        float maf = sigm(tmisca[c] + matmp[i]);
        float mkf = sigm(tmisck[c] + mktmp[i]);
        float mvf = sigm(tmiscv[c] + mvtmp[i]);
        float kr  = kraw_k[i];
        float vr  = vraw_v[i];
        oa[q]   = a;
        ov[q]   = vr + (v1[i] - vr) * mvf;
        kkun[q] = kr + kktmp_kk[i];
        float kn = kr * maf + kr * a * (1.0f - maf);
        kn *= expf(fminf(wv * mkf, 0.0f));
        ok[q] = kn;
    }
    float ss  = warp_sum(kkun[0] * kkun[0] + kkun[1] * kkun[1]);
    float nrm = fmaxf(sqrtf(ss), 1e-12f);
    #pragma unroll
    for (int q = 0; q < 2; ++q) {
        size_t i = ii[q];
        wtmp_d[i]   = od[q];
        atmp_a[i]   = oa[q];
        kktmp_kk[i] = kkun[q] / nrm;
        kraw_k[i]   = ok[q];
        vraw_v[i]   = ov[q];
    }
}

// RWKV7 delta-rule recurrence. One block per (b,h); 256 threads hold the
// 64x64 fp32 state in registers (row = tid/4, 16-col group per thread).
__global__ __launch_bounds__(256)
void rwkv7_scan(const float* __restrict__ r, const float* __restrict__ d,
                const float* __restrict__ k, const float* __restrict__ v,
                const float* __restrict__ kk, const float* __restrict__ asig,
                float* __restrict__ y)
{
    int bh = blockIdx.x;
    int b  = bh / kH;
    int h  = bh % kH;
    int tid = threadIdx.x;
    int row = tid >> 2;
    int grp = tid & 3;
    int j0  = grp * 16;

    __shared__ float sr[kN], sd[kN], sk[kN], sv[kN], saa[kN], sbb[kN];
    __shared__ float pA[kN][4];
    __shared__ float pB[kN][4];

    float S[16];
    #pragma unroll
    for (int j = 0; j < 16; ++j) S[j] = 0.0f;

    for (int t = 0; t < kT; ++t) {
        size_t base = ((size_t)(b * kT + t)) * kC + h * kN;
        if (tid < kN) {
            float kkv = kk[base + tid];
            float av  = asig[base + tid];
            sr[tid]  = r[base + tid];
            sd[tid]  = d[base + tid];
            sk[tid]  = k[base + tid];
            sv[tid]  = v[base + tid];
            saa[tid] = -kkv;
            sbb[tid] = kkv * av;
        }
        __syncthreads();

        float ps = 0.0f;
        #pragma unroll
        for (int jj = 0; jj < 16; ++jj) ps += S[jj] * saa[j0 + jj];
        pA[row][grp] = ps;
        __syncthreads();

        float sa = pA[row][0] + pA[row][1] + pA[row][2] + pA[row][3];
        float vi = sv[row];
        float pr = 0.0f;
        #pragma unroll
        for (int jj = 0; jj < 16; ++jj) {
            int j = j0 + jj;
            float s = S[jj] * sd[j] + sa * sbb[j] + vi * sk[j];
            S[jj] = s;
            pr += s * sr[j];
        }
        pB[row][grp] = pr;
        __syncthreads();

        if (grp == 0)
            y[base + row] = pB[row][0] + pB[row][1] + pB[row][2] + pB[row][3];
    }
}

// GroupNorm + bonus + gate: one warp per head.
__global__ __launch_bounds__(384)
void post_kernel(const float* __restrict__ y, const float* __restrict__ r,
                 const float* __restrict__ k, const float* __restrict__ v,
                 const float* __restrict__ g, const float* __restrict__ faaaa,
                 const float* __restrict__ gn_w, const float* __restrict__ gn_b,
                 float* __restrict__ yg)
{
    int bt = blockIdx.x;
    int h  = threadIdx.x >> 5;
    int l  = threadIdx.x & 31;
    size_t base = (size_t)bt * kC + h * kN;
    size_t i0 = base + l, i1 = i0 + 32;
    int c0 = h * kN + l, c1 = c0 + 32;

    float y0 = y[i0], y1 = y[i1];
    float mean = warp_sum(y0 + y1) * (1.0f / kN);
    float d0 = y0 - mean, d1 = y1 - mean;
    float var = warp_sum(d0 * d0 + d1 * d1) * (1.0f / kN);
    float inv = rsqrtf(var + kGnEps);
    float n0 = d0 * inv * gn_w[c0] + gn_b[c0];
    float n1 = d1 * inv * gn_w[c1] + gn_b[c1];

    float rk = r[i0] * k[i0] * faaaa[c0] + r[i1] * k[i1] * faaaa[c1];
    float dot = warp_sum(rk);

    yg[i0] = (n0 + dot * v[i0]) * g[i0];
    yg[i1] = (n1 + dot * v[i1]) * g[i1];
}

// ---------------- host-side orchestration ----------------
static inline void gemm(const float* A, int lda, const float* Bt, int ldbt,
                        float* C, int ldc, int M, int N, int K, int act,
                        hipStream_t s)
{
    dim3 grid(N / GBN, M / GBM);
    gemm_wmma_bf16<<<grid, 256, 0, s>>>(A, lda, Bt, ldbt, C, ldc, M, N, K, act);
}

static inline void tpad(const float* src, int K0, int N0,
                        float* dst, int Np, int Kp, hipStream_t s)
{
    transpose_pad_kernel<<<(Np * Kp + 255) / 256, 256, 0, s>>>(src, K0, N0,
                                                               dst, Np, Kp);
}

extern "C" void kernel_launch(void* const* d_in, const int* in_sizes, int n_in,
                              void* d_out, int out_size, void* d_ws, size_t ws_size,
                              hipStream_t stream)
{
    const float* x       = (const float*)d_in[0];
    const float* v1      = (const float*)d_in[1];
    const float* tmx     = (const float*)d_in[2];
    const float* tmrg    = (const float*)d_in[3];
    const float* tmwa    = (const float*)d_in[4];
    const float* tmk     = (const float*)d_in[5];
    const float* tmv     = (const float*)d_in[6];
    const float* tdecay  = (const float*)d_in[7];
    const float* tfaaaa  = (const float*)d_in[8];
    const float* taaaaa  = (const float*)d_in[9];
    const float* maa_w1  = (const float*)d_in[10];
    const float* maa_w2  = (const float*)d_in[11];
    const float* decay_w1= (const float*)d_in[12];
    const float* decay_w2= (const float*)d_in[13];
    const float* aaa_w1  = (const float*)d_in[14];
    const float* aaa_w2  = (const float*)d_in[15];
    const float* kkk_w1  = (const float*)d_in[16];
    const float* kkk_w2  = (const float*)d_in[17];
    const float* gate_w1 = (const float*)d_in[18];
    const float* gate_w2 = (const float*)d_in[19];
    const float* ma_w1   = (const float*)d_in[20];
    const float* ma_w2   = (const float*)d_in[21];
    const float* tmisca  = (const float*)d_in[22];
    const float* mk_w1   = (const float*)d_in[23];
    const float* mk_w2   = (const float*)d_in[24];
    const float* tmisck  = (const float*)d_in[25];
    const float* mv_w1   = (const float*)d_in[26];
    const float* mv_w2   = (const float*)d_in[27];
    const float* tmiscv  = (const float*)d_in[28];
    const float* Wr      = (const float*)d_in[29];
    const float* Wk      = (const float*)d_in[30];
    const float* Wv      = (const float*)d_in[31];
    const float* Wo      = (const float*)d_in[32];
    const float* gn_w    = (const float*)d_in[33];
    const float* gn_b    = (const float*)d_in[34];

    float* out = (float*)d_out;
    float* ws  = (float*)d_ws;

    const size_t PL = (size_t)kBT * kC;           // one (B,T,C) plane
    float* s0  = ws + 0 * PL;   // xx      -> atmp  -> a
    float* s1  = ws + 1 * PL;   // xxx     -> kktmp -> kk
    float* s2  = ws + 2 * PL;   // adds0   -> r
    float* s3  = ws + 3 * PL;   // adds1   -> kraw  -> k
    float* s4  = ws + 4 * PL;   // adds2   -> vraw  -> v
    float* s5  = ws + 5 * PL;   // adds3   -> wtmp  -> d
    float* s6  = ws + 6 * PL;   // xrg     -> mktmp -> y
    float* s7  = ws + 7 * PL;   // xwa     -> yg
    float* s8  = ws + 8 * PL;   // xk      -> mvtmp
    float* s9  = ws + 9 * PL;   // xv      -> matmp
    float* s10 = ws + 10 * PL;  // g
    float* p = ws + 11 * PL;
    float* mmb   = p; p += (size_t)kBT * 128;     // 4096 x 128
    float* t4b   = p; p += (size_t)kBT * 128;     // 4096 x 128
    float* t1b   = p; p += (size_t)kBT * 64;      // 4096 x 64
    float* tsb   = p; p += (size_t)kBT * 64;      // 4096 x 64
    // transposed / padded weights (Bt form: N x K)
    float* maw1t = p; p += (size_t)128 * kC;      // 128 x 768
    float* maw2t = p; p += (size_t)4 * kC * 32;   // 4 x (768 x 32)
    float* gw1t  = p; p += (size_t)128 * kC;      // 128 x 768
    float* gw2t  = p; p += (size_t)kC * 128;      // 768 x 128
    float* dw1t  = p; p += (size_t)64 * kC;       // 64 x 768
    float* dw2t  = p; p += (size_t)kC * 64;       // 768 x 64
    float* w1t[5]; const float* w1s[5] = { aaa_w1, kkk_w1, ma_w1, mk_w1, mv_w1 };
    float* w2t[5]; const float* w2s[5] = { aaa_w2, kkk_w2, ma_w2, mk_w2, mv_w2 };
    for (int i = 0; i < 5; ++i) { w1t[i] = p; p += (size_t)64 * kC; }
    for (int i = 0; i < 5; ++i) { w2t[i] = p; p += (size_t)kC * 64; }
    float* WrT = p; p += (size_t)kC * kC;
    float* WkT = p; p += (size_t)kC * kC;
    float* WvT = p; p += (size_t)kC * kC;
    float* WoT = p; p += (size_t)kC * kC;

    const int total = kBT * kC;
    const int ngrid = (total + 255) / 256;

    // 0) transpose + zero-pad all weights into Bt (N x K) form
    trans_maa_w1_kernel<<<(128 * kC + 255) / 256, 256, 0, stream>>>(maa_w1, maw1t);
    for (int f = 0; f < 4; ++f)
        tpad(maa_w2 + (size_t)f * 28 * kC, 28, kC, maw2t + (size_t)f * kC * 32,
             kC, 32, stream);
    tpad(gate_w1, kC, 120, gw1t, 128, kC, stream);
    tpad(gate_w2, 120, kC, gw2t, kC, 128, stream);
    tpad(decay_w1, kC, 64, dw1t, 64, kC, stream);
    tpad(decay_w2, 64, kC, dw2t, kC, 64, stream);
    for (int i = 0; i < 5; ++i) {
        tpad(w1s[i], kC, 16, w1t[i], 64, kC, stream);
        tpad(w2s[i], 16, kC, w2t[i], kC, 64, stream);
    }
    tpad(Wr, kC, kC, WrT, kC, kC, stream);
    tpad(Wk, kC, kC, WkT, kC, kC, stream);
    tpad(Wv, kC, kC, WvT, kC, kC, stream);
    tpad(Wo, kC, kC, WoT, kC, kC, stream);

    // 1) token shift + maa_x mix
    prelude_kernel<<<ngrid, 256, 0, stream>>>(x, tmx, s0, s1, total);

    // 2) mm = tanh(xxx @ maa_w1)  [4096 x 128], chunk f at cols 32f..
    gemm(s1, kC, maw1t, kC, mmb, 128, kBT, 128, kC, 1, stream);

    // 3) adds_f = mm[:, 32f:32f+32] @ maa_w2[f]  [4096 x 768]
    float* addsl[4] = { s2, s3, s4, s5 };
    for (int f = 0; f < 4; ++f)
        gemm(mmb + 32 * f, 128, maw2t + (size_t)f * kC * 32, 32,
             addsl[f], kC, kBT, kC, 32, 0, stream);

    // 4) xrg/xwa/xk/xv
    mix4_kernel<<<ngrid, 256, 0, stream>>>(x, s0, s2, s3, s4, s5,
                                           tmrg, tmwa, tmk, tmv,
                                           s6, s7, s8, s9, total);

    // 5) r = xrg @ Wr
    gemm(s6, kC, WrT, kC, s2, kC, kBT, kC, kC, 0, stream);
    // 6) g = sigmoid(xrg @ gate_w1) @ gate_w2
    gemm(s6, kC, gw1t, kC, t4b, 128, kBT, 128, kC, 2, stream);
    gemm(t4b, 128, gw2t, 128, s10, kC, kBT, kC, 128, 0, stream);
    // 7) kraw = xk @ Wk
    gemm(s8, kC, WkT, kC, s3, kC, kBT, kC, kC, 0, stream);
    // 8) kktmp = tanh(xk @ kkk_w1) @ kkk_w2
    gemm(s8, kC, w1t[1], kC, tsb, 64, kBT, 64, kC, 1, stream);
    gemm(tsb, 64, w2t[1], 64, s1, kC, kBT, kC, 64, 0, stream);
    // 9) mktmp = (xk @ mk_w1) @ mk_w2
    gemm(s8, kC, w1t[3], kC, tsb, 64, kBT, 64, kC, 0, stream);
    gemm(tsb, 64, w2t[3], 64, s6, kC, kBT, kC, 64, 0, stream);
    // 10) vraw = xv @ Wv
    gemm(s9, kC, WvT, kC, s4, kC, kBT, kC, kC, 0, stream);
    // 11) mvtmp = (xv @ mv_w1) @ mv_w2
    gemm(s9, kC, w1t[4], kC, tsb, 64, kBT, 64, kC, 0, stream);
    gemm(tsb, 64, w2t[4], 64, s8, kC, kBT, kC, 64, 0, stream);
    // 12) wtmp = tanh(xwa @ decay_w1) @ decay_w2
    gemm(s7, kC, dw1t, kC, t1b, 64, kBT, 64, kC, 1, stream);
    gemm(t1b, 64, dw2t, 64, s5, kC, kBT, kC, 64, 0, stream);
    // 13) atmp = (xwa @ aaa_w1) @ aaa_w2
    gemm(s7, kC, w1t[0], kC, tsb, 64, kBT, 64, kC, 0, stream);
    gemm(tsb, 64, w2t[0], 64, s0, kC, kBT, kC, 64, 0, stream);
    // 14) matmp = (xwa @ ma_w1) @ ma_w2
    gemm(s7, kC, w1t[2], kC, tsb, 64, kBT, 64, kC, 0, stream);
    gemm(tsb, 64, w2t[2], 64, s9, kC, kBT, kC, 64, 0, stream);

    // 15) fused nonlinearities + head-norm (in-place: d,a,kk,k,v)
    fuse_pre_scan<<<kBT, 384, 0, stream>>>(s5, s0, s1, s9, s6, s8, s3, s4,
                                           v1, tdecay, taaaaa, tmisca,
                                           tmisck, tmiscv);

    // 16) delta-rule recurrence -> y (s6)
    rwkv7_scan<<<kB * kH, 256, 0, stream>>>(s2, s5, s3, s4, s1, s0, s6);

    // 17) groupnorm + bonus + gate -> yg (s7)
    post_kernel<<<kBT, 384, 0, stream>>>(s6, s2, s3, s4, s10, tfaaaa,
                                         gn_w, gn_b, s7);

    // 18) out = yg @ Wo
    gemm(s7, kC, WoT, kC, out, kC, kBT, kC, kC, 0, stream);

    // 19) second tuple output: v1 passthrough
    hipMemcpyAsync(out + (size_t)kBT * kC, (void*)v1,
                   (size_t)kBT * kC * sizeof(float),
                   hipMemcpyDeviceToDevice, stream);
}
